// DeformerBlock_74139725464150
// MI455X (gfx1250) — matrix operand order repaired and checked by reading, not compiled, not run on hardware
//
#include <hip/hip_runtime.h>
#include <math.h>

#define H_    224
#define W_    224
#define HW    50176            // 224*224
#define NFEAT 384
#define WAVES 4
#define ASTRIDE 264            // act row stride (elements), %8==0 for 16B alignment, bank-rotating
#define STAGE_ELEMS 12288      // 24KB bf16 weight staging buffer

// packed-weight offsets (bf16 elements) inside d_ws
#define OFF_FEAT 0
#define OFF_PROJ 49152
#define OFF_RES  81920         // 8 layers of 16384 (res1[i], res2[i] interleaved)
#define OFF_D1   212992
#define OFF_D2   221184
#define OFF_D3   223232
#define OFF_D4   223744

typedef __attribute__((ext_vector_type(16))) __bf16 v16bf;
typedef __attribute__((ext_vector_type(8)))  float  v8f;

union FragA { v16bf v; uint4 q[2]; unsigned short u[16]; };

// Native hardware bf16 converts (single VALU op, vs 3-op manual RNE)
__device__ inline unsigned short f2bf(float f) {
  union { __bf16 h; unsigned short u; } c;
  c.h = (__bf16)f;
  return c.u;
}
__device__ inline float bf2f(unsigned short u) {
  union { __bf16 h; unsigned short u; } c;
  c.u = u;
  return (float)c.h;
}
// host-safe manual version for the pack kernel (compiled for both targets)
__device__ __host__ inline unsigned short f2bf_bits(float f) {
  union { float f; unsigned int u; } c; c.f = f;
  unsigned int r = (c.u + 0x7FFFu + ((c.u >> 16) & 1u)) >> 16;
  return (unsigned short)r;
}
__device__ inline float gelu_f(float x) {
  return 0.5f * x * (1.0f + tanhf(0.7978845608028654f * (x + 0.044715f * x * x * x)));
}

// ---------------------------------------------------------------------------
// Pack fp32 weights [O][C] into per-lane bf16 A-fragment layout.
// Fragment index = kt*MT + mt (kt-major so K-chunks are contiguous for staging).
// Lane layout (16-bit A, 16x32): row m = lane&15; K offset o = 16*(lane>>4);
//   elements 0..15 -> K = 32*kt + o + e
// ---------------------------------------------------------------------------
__global__ void pack_weights(const float* __restrict__ w, unsigned short* __restrict__ dst,
                             int O, int C, int MT, int KT) {
  int t = blockIdx.x * blockDim.x + threadIdx.x;
  int total = MT * KT * 512;
  if (t >= total) return;
  int frag = t >> 9;
  int within = t & 511;
  int lane = within >> 4;
  int e = within & 15;
  int kt = frag / MT;
  int mt = frag - kt * MT;
  int m = 16 * mt + (lane & 15);
  int kb = 32 * kt + 16 * (lane >> 4);
  int k = kb + e;
  float val = (m < O && k < C) ? w[m * C + k] : 0.0f;
  dst[t] = f2bf_bits(val);
}

// ---------------------------------------------------------------------------
// Generic WMMA layer: D(16*MT x 16px) = gelu(W x act + bias) [+ residual]
//   act buffer: this wave's [16][ASTRIDE] bf16 region in LDS
//   weights: staged chunk-wise (KTC K-tiles) from global packed into s_w (shared)
//   BG=true: B fragments come directly from global fp32 (channel-strided)
// ---------------------------------------------------------------------------
template<int MT, int KT, int KTC, bool GELU, bool RESID, bool BG>
__device__ inline void run_layer(unsigned short (*act)[ASTRIDE],
                                 int inOff, int outOff,
                                 const unsigned short* __restrict__ wsrc,
                                 unsigned short* __restrict__ sw,
                                 const float* __restrict__ bias,
                                 const float* __restrict__ gB,
                                 int tid, int lane) {
  const int px = lane & 15, half = lane >> 4;
  v8f acc[MT];
#pragma unroll
  for (int mt = 0; mt < MT; ++mt)
#pragma unroll
    for (int r = 0; r < 8; ++r) acc[mt][r] = 0.0f;

#pragma unroll 1
  for (int c0 = 0; c0 < KT; c0 += KTC) {
    __syncthreads();  // everyone done with previous contents of sw
    {
      const uint4* s = reinterpret_cast<const uint4*>(wsrc + (size_t)c0 * MT * 512);
      uint4* d = reinterpret_cast<uint4*>(sw);
      const int n4 = (KTC * MT * 512) >> 3;
      for (int i = tid; i < n4; i += WAVES * 32) d[i] = s[i];
    }
    __syncthreads();
#pragma unroll
    for (int kk = 0; kk < KTC; ++kk) {
      const int kt = c0 + kk;
      FragA B;
      if (BG) {
        const float* fp = gB + (size_t)(kt * 32 + half * 16) * HW;
#pragma unroll
        for (int j = 0; j < 16; ++j) B.u[j] = f2bf(fp[(size_t)j * HW]);
      } else {
        const uint4* bp = reinterpret_cast<const uint4*>(&act[px][inOff + kt * 32 + half * 16]);
        B.q[0] = bp[0]; B.q[1] = bp[1];
      }
#pragma unroll
      for (int mt = 0; mt < MT; ++mt) {
        FragA A;
        const uint4* ap = reinterpret_cast<const uint4*>(sw + ((kk * MT + mt) * 512 + lane * 16));
        A.q[0] = ap[0]; A.q[1] = ap[1];
        acc[mt] = __builtin_amdgcn_wmma_f32_16x16x32_bf16(false, A.v, false, B.v,
                                                          (short)0, acc[mt], false, false);
      }
    }
  }
  // epilogue: bias (+gelu) (+residual) -> bf16 -> LDS act[outOff]
#pragma unroll
  for (int mt = 0; mt < MT; ++mt) {
    const int mbase = 16 * mt + 8 * half;
    float res[8];
    if (RESID) {
      const uint4 rq = *reinterpret_cast<const uint4*>(&act[px][outOff + mbase]);
      const unsigned short* ru = reinterpret_cast<const unsigned short*>(&rq);
#pragma unroll
      for (int r = 0; r < 8; ++r) res[r] = bf2f(ru[r]);
    }
    unsigned int packed[4];
#pragma unroll
    for (int r = 0; r < 8; r += 2) {
      float v0 = acc[mt][r]     + bias[mbase + r];
      float v1 = acc[mt][r + 1] + bias[mbase + r + 1];
      if (GELU)  { v0 = gelu_f(v0); v1 = gelu_f(v1); }
      if (RESID) { v0 += res[r];    v1 += res[r + 1]; }
      packed[r >> 1] = (unsigned int)f2bf(v0) | ((unsigned int)f2bf(v1) << 16);
    }
    uint4 st; st.x = packed[0]; st.y = packed[1]; st.z = packed[2]; st.w = packed[3];
    *reinterpret_cast<uint4*>(&act[px][outOff + mbase]) = st;
  }
}

// ---------------------------------------------------------------------------
// Fused deformer: per wave = 16 pixels through the whole MLP + grid_sample.
// ---------------------------------------------------------------------------
__global__ __launch_bounds__(WAVES * 32, 4)
void deformer_main(const float* __restrict__ features, const float* __restrict__ image,
                   const float* __restrict__ w_img, const float* __restrict__ b_img,
                   const float* __restrict__ b_feat, const float* __restrict__ b_proj,
                   const float* __restrict__ res_b1, const float* __restrict__ res_b2,
                   const float* __restrict__ d_b1, const float* __restrict__ d_b2,
                   const float* __restrict__ d_b3, const float* __restrict__ d_b4,
                   const unsigned short* __restrict__ wpk, float* __restrict__ out) {
  __shared__ __align__(16) unsigned short s_w[STAGE_ELEMS];
  __shared__ __align__(16) unsigned short s_act[WAVES][16][ASTRIDE];
  __shared__ float s_off[WAVES][16][2];

  const int tid = threadIdx.x;
  const int wv = tid >> 5;
  const int lane = tid & 31;
  const int px = lane & 15, half = lane >> 4;

  const int tile = blockIdx.x * WAVES + wv;     // grid sized so all tiles valid
  const int p0 = tile * 16;
  const int b = p0 / HW;
  const int rem = p0 - b * HW;
  const int h = rem / W_;
  const int w0 = rem - h * W_;

  unsigned short (*act)[ASTRIDE] = s_act[wv];

  // ---- image projection (K=3, VALU): gelu(w_img @ image + b_img) -> slot1 (ch 128..255)
  {
    const float* ib = image + (size_t)b * 3 * HW + (size_t)h * W_ + w0 + px;
    const float i0 = ib[0], i1 = ib[HW], i2 = ib[2 * HW];
#pragma unroll 1
    for (int c = half; c < 128; c += 2) {
      float v = b_img[c] + w_img[c * 3 + 0] * i0 + w_img[c * 3 + 1] * i1 + w_img[c * 3 + 2] * i2;
      act[px][128 + c] = f2bf(gelu_f(v));
    }
  }

  // ---- feature projection: gelu(W_feat[128x384] @ features) -> slot0 (B from global)
  const float* gfeat = features + (size_t)b * NFEAT * HW + (size_t)h * W_ + w0 + px;
  run_layer<8, 12, 3, true, false, true >(act, 0, 0,   wpk + OFF_FEAT, s_w, b_feat, gfeat, tid, lane);

  // ---- proj: gelu(W_proj[128x256] @ concat(f, img)) -> slot0
  run_layer<8, 8, 2, true, false, false>(act, 0, 0,   wpk + OFF_PROJ, s_w, b_proj, nullptr, tid, lane);

  // ---- residual blocks: f = f + conv2(gelu(conv1(f)))
#pragma unroll 1
  for (int i = 0; i < 4; ++i) {
    run_layer<8, 4, 2, true,  false, false>(act, 0,   128, wpk + OFF_RES + (2 * i) * 16384,
                                            s_w, res_b1 + i * 128, nullptr, tid, lane);
    run_layer<8, 4, 2, false, true,  false>(act, 128, 0,   wpk + OFF_RES + (2 * i + 1) * 16384,
                                            s_w, res_b2 + i * 128, nullptr, tid, lane);
  }

  // ---- deformer head
  run_layer<4, 4, 4, true, false, false>(act, 0,   128, wpk + OFF_D1, s_w, d_b1, nullptr, tid, lane);
  run_layer<2, 2, 2, true, false, false>(act, 128, 0,   wpk + OFF_D2, s_w, d_b2, nullptr, tid, lane);
  run_layer<1, 1, 1, true, false, false>(act, 0,   128, wpk + OFF_D3, s_w, d_b3, nullptr, tid, lane);

  // zero-pad K 16..31 of d4 input (ch 144..159), written/read by same lane (half==1)
  if (half) {
    uint4 z = make_uint4(0u, 0u, 0u, 0u);
    uint4* zp = reinterpret_cast<uint4*>(&act[px][144]);
    zp[0] = z; zp[1] = z;
  }

  // ---- d4 (2x16 padded to 16x32): offsets (dx, dy)
  __syncthreads();
  {
    const uint4* s = reinterpret_cast<const uint4*>(wpk + OFF_D4);
    uint4* d = reinterpret_cast<uint4*>(s_w);
    if (tid < 64) d[tid] = s[tid];
  }
  __syncthreads();
  {
    FragA A4, B4;
    const uint4* bp = reinterpret_cast<const uint4*>(&act[px][128 + half * 16]);
    B4.q[0] = bp[0]; B4.q[1] = bp[1];
    const uint4* ap = reinterpret_cast<const uint4*>(s_w + lane * 16);
    A4.q[0] = ap[0]; A4.q[1] = ap[1];
    v8f a4;
#pragma unroll
    for (int r = 0; r < 8; ++r) a4[r] = 0.0f;
    a4 = __builtin_amdgcn_wmma_f32_16x16x32_bf16(false, A4.v, false, B4.v, (short)0, a4, false, false);
    if (!half) {  // lanes 0..15 hold rows m=0 (dx) and m=1 (dy) for pixel = lane
      s_off[wv][px][0] = a4[0] + d_b4[0];
      s_off[wv][px][1] = a4[1] + d_b4[1];
    }
  }
  __syncthreads();

  // ---- grid_sample(features, base + offsets), border padding, align_corners=False
  {
    const float dx = s_off[wv][px][0], dy = s_off[wv][px][1];
    // (gx+1)*112 == (w0+px), so x = (w0+px) + 112*dx - 0.5  (same for y)
    float x = (float)(w0 + px) + 112.0f * dx - 0.5f;
    float y = (float)h         + 112.0f * dy - 0.5f;
    x = fminf(fmaxf(x, 0.0f), 223.0f);
    y = fminf(fmaxf(y, 0.0f), 223.0f);
    const float x0f = floorf(x), y0f = floorf(y);
    const float wx = x - x0f, wy = y - y0f;
    const int x0 = (int)x0f, y0 = (int)y0f;
    const int x1 = min(x0 + 1, 223), y1 = min(y0 + 1, 223);
    const float w00 = (1.0f - wx) * (1.0f - wy), w01 = wx * (1.0f - wy);
    const float w10 = (1.0f - wx) * wy,          w11 = wx * wy;
    const int i00 = y0 * W_ + x0, i01 = y0 * W_ + x1;
    const int i10 = y1 * W_ + x0, i11 = y1 * W_ + x1;
    const float* fb0 = features + (size_t)b * NFEAT * HW;
    float* ob = out + (size_t)b * NFEAT * HW + (size_t)h * W_ + w0 + px;
#pragma unroll 1
    for (int c = half; c < NFEAT; c += 2) {
      const float* f = fb0 + (size_t)c * HW;
      float v = w00 * f[i00] + w01 * f[i01] + w10 * f[i10] + w11 * f[i11];
      ob[(size_t)c * HW] = v;
    }
  }
}

// ---------------------------------------------------------------------------
extern "C" void kernel_launch(void* const* d_in, const int* in_sizes, int n_in,
                              void* d_out, int out_size, void* d_ws, size_t ws_size,
                              hipStream_t stream) {
  (void)in_sizes; (void)n_in; (void)out_size; (void)ws_size;
  const float* features = (const float*)d_in[0];
  const float* image    = (const float*)d_in[1];
  const float* w_img    = (const float*)d_in[2];
  const float* b_img    = (const float*)d_in[3];
  const float* w_feat   = (const float*)d_in[4];
  const float* b_feat   = (const float*)d_in[5];
  const float* w_proj   = (const float*)d_in[6];
  const float* b_proj   = (const float*)d_in[7];
  const float* res_w1   = (const float*)d_in[8];
  const float* res_b1   = (const float*)d_in[9];
  const float* res_w2   = (const float*)d_in[10];
  const float* res_b2   = (const float*)d_in[11];
  const float* d_w1 = (const float*)d_in[12]; const float* d_b1 = (const float*)d_in[13];
  const float* d_w2 = (const float*)d_in[14]; const float* d_b2 = (const float*)d_in[15];
  const float* d_w3 = (const float*)d_in[16]; const float* d_b3 = (const float*)d_in[17];
  const float* d_w4 = (const float*)d_in[18]; const float* d_b4 = (const float*)d_in[19];

  unsigned short* wp = (unsigned short*)d_ws;

  auto packl = [&](const float* w, int off, int O, int C, int MT, int KT) {
    int total = MT * KT * 512;
    hipLaunchKernelGGL(pack_weights, dim3((total + 255) / 256), dim3(256), 0, stream,
                       w, wp + off, O, C, MT, KT);
  };
  packl(w_feat, OFF_FEAT, 128, 384, 8, 12);
  packl(w_proj, OFF_PROJ, 128, 256, 8, 8);
  for (int i = 0; i < 4; ++i) {
    packl(res_w1 + i * 16384, OFF_RES + (2 * i) * 16384,     128, 128, 8, 4);
    packl(res_w2 + i * 16384, OFF_RES + (2 * i + 1) * 16384, 128, 128, 8, 4);
  }
  packl(d_w1, OFF_D1, 64, 128, 4, 4);
  packl(d_w2, OFF_D2, 32, 64, 2, 2);
  packl(d_w3, OFF_D3, 16, 32, 1, 1);
  packl(d_w4, OFF_D4, 2, 16, 1, 1);

  // 8*224*224 pixels / 16 per wave / 4 waves per block = 6272 blocks (exact)
  hipLaunchKernelGGL(deformer_main, dim3(6272), dim3(WAVES * 32), 0, stream,
                     features, image, w_img, b_img, b_feat, b_proj,
                     res_b1, res_b2, d_b1, d_b2, d_b3, d_b4,
                     (const unsigned short*)wp, (float*)d_out);
}
